// MultiHeadAttention_50311246905858
// MI455X (gfx1250) — matrix operand order repaired and checked by reading, not compiled, not run on hardware
//
#include <hip/hip_runtime.h>
#include <stdint.h>

#define D_MODEL 1024
#define SEQ     2048
#define BATCH   2
#define NH      16
#define DKH     64
#define MTOT    (BATCH * SEQ)   // 4096

typedef __attribute__((ext_vector_type(16))) __bf16          v16bf;
typedef __attribute__((ext_vector_type(8)))  float           v8f;
typedef __attribute__((ext_vector_type(8)))  unsigned short  ushort8;

union Frag16 { v16bf v; ushort8 h[2]; };

__device__ __forceinline__ unsigned short f2bf(float f) {
  unsigned int u = __builtin_bit_cast(unsigned int, f);
  u += 0x7FFFu + ((u >> 16) & 1u);          // round-to-nearest-even
  return (unsigned short)(u >> 16);
}

__device__ __forceinline__ v8f v8f_zero() {
  v8f z = {0.f, 0.f, 0.f, 0.f, 0.f, 0.f, 0.f, 0.f};
  return z;
}

// ---------------- elementwise f32 -> bf16 ----------------
__global__ void cvt_f32_to_bf16(const float* __restrict__ in,
                                unsigned short* __restrict__ out, int n) {
  int i = blockIdx.x * blockDim.x + threadIdx.x;
  if (i < n) out[i] = f2bf(in[i]);
}

// ---- W [K][N] f32 -> W^T [N][K] bf16 (rows of W^T are contiguous K) ----
__global__ void transpose_cvt(const float* __restrict__ in,
                              unsigned short* __restrict__ out) {
  int i = blockIdx.x * blockDim.x + threadIdx.x;   // over N*K = 1M
  int n = i >> 10;
  int k = i & 1023;
  out[i] = f2bf(in[k * D_MODEL + n]);
}

// =====================================================================
// bf16 WMMA GEMM: C[M=4096][N=1024] = A[M][1024] * B[1024][N] + bias
// BT is B transposed [N][K] bf16. One wave -> 16x64 tile, K step 32.
// B tile (32K x 64N) staged in double-buffered LDS, shared by 8 waves.
// MODE 0: out fp32 row-major (final projection -> d_out)
// MODE 1: Q  -> bf16 [b][h][s][d], scaled by log2(e)/sqrt(dk)
// MODE 2: K  -> bf16 [b][h][s][d]
// MODE 3: V  -> bf16 [b][h][d][s]  (transposed for PV A-fragments)
// =====================================================================
template <int MODE>
__global__ void gemm_wmma(const unsigned short* __restrict__ A,
                          const unsigned short* __restrict__ BT,
                          const float* __restrict__ bias,
                          float* __restrict__ outF,
                          unsigned short* __restrict__ outB) {
  __shared__ unsigned short Bl[2][64 * 32];        // 2 x 4KB K-step tiles
  const int tid    = threadIdx.x;
  const int lane   = tid & 31;
  const int wave   = tid >> 5;
  const int half16 = (lane >= 16) ? 1 : 0;
  const int lr     = lane & 15;
  const int m0 = blockIdx.y * 128 + wave * 16;
  const int n0 = blockIdx.x * 64;

  const int sn = tid >> 2;                         // 0..63
  const int sk = (tid & 3) * 8;                    // 0,8,16,24
  const unsigned short* bstage = BT + (size_t)(n0 + sn) * D_MODEL + sk;

  v8f acc[4];
#pragma unroll
  for (int t = 0; t < 4; ++t) acc[t] = v8f_zero();

  const unsigned short* aptr = A + (size_t)(m0 + lr) * D_MODEL;

  *(ushort8*)(&Bl[0][sn * 32 + sk]) = *(const ushort8*)(bstage);

  int cur = 0;
  for (int kk = 0; kk < D_MODEL; kk += 32) {
    __syncthreads();
    if (kk + 32 < D_MODEL)
      *(ushort8*)(&Bl[cur ^ 1][sn * 32 + sk]) =
          *(const ushort8*)(bstage + kk + 32);
    __builtin_prefetch(aptr + kk + 256, 0, 0);

    Frag16 a;                                       // A 16x32 fragment
    const int ak = kk + (half16 ? 8 : 0);
    a.h[0] = *(const ushort8*)(aptr + ak);
    a.h[1] = *(const ushort8*)(aptr + ak + 16);

    const unsigned short* bl = &Bl[cur][0];
#pragma unroll
    for (int t = 0; t < 4; ++t) {                   // B 32x16 fragments (LDS)
      Frag16 bfr;
      const int nl = t * 16 + lr;
      const int kb = half16 ? 8 : 0;
      bfr.h[0] = *(const ushort8*)(bl + nl * 32 + kb);
      bfr.h[1] = *(const ushort8*)(bl + nl * 32 + kb + 16);
      acc[t] = __builtin_amdgcn_wmma_f32_16x16x32_bf16(
          false, a.v, false, bfr.v, (short)0, acc[t], false, false);
    }
    cur ^= 1;
  }

#pragma unroll
  for (int t = 0; t < 4; ++t) {
    const int n  = n0 + t * 16 + lr;
    const float bvv = bias[n];
#pragma unroll
    for (int r = 0; r < 8; ++r) {
      const int m = m0 + r + half16 * 8;            // C/D: lane=N, vgpr(+half)=M
      float v = acc[t][r] + bvv;
      if (MODE == 0) {
        outF[(size_t)m * D_MODEL + n] = v;
      } else {
        const int b = m >> 11, s = m & 2047;
        const int h = n >> 6,  d = n & 63;
        if (MODE == 1) v *= 0.18033688f;            // log2(e)/sqrt(64) folded
        const unsigned short q = f2bf(v);
        if (MODE == 3)
          outB[(((size_t)(b * NH + h)) * DKH + d) * SEQ + s] = q;
        else
          outB[(((size_t)(b * NH + h)) * SEQ + s) * DKH + d] = q;
      }
    }
  }
}

// =====================================================================
// One 32-key block of transposed flash attention.
// MASKED=false: block fully below the causal diagonal (no compares).
// =====================================================================
template <bool MASKED>
__device__ __forceinline__ void attn_block(const unsigned short* __restrict__ Kh,
                                           const unsigned short* __restrict__ Vh,
                                           const Frag16 (&bq_)[2],
                                           v8f (&ot)[4],
                                           float& m, float& l,
                                           int kk, int query, int half16, int lr) {
  // ---- S^T tiles: sta = keys kk..kk+15, stb = keys kk+16..kk+31
  v8f sta = v8f_zero(), stb = v8f_zero();
#pragma unroll
  for (int j = 0; j < 2; ++j) {
    Frag16 ka, kb2;
    const int dkb = 32 * j + (half16 ? 8 : 0);     // A-frag dk pattern
    const unsigned short* k0 = Kh + (size_t)(kk + lr) * DKH + dkb;
    const unsigned short* k1 = Kh + (size_t)(kk + 16 + lr) * DKH + dkb;
    ka.h[0]  = *(const ushort8*)(k0); ka.h[1]  = *(const ushort8*)(k0 + 16);
    kb2.h[0] = *(const ushort8*)(k1); kb2.h[1] = *(const ushort8*)(k1 + 16);
    sta = __builtin_amdgcn_wmma_f32_16x16x32_bf16(false, ka.v,  false, bq_[j].v,
                                                  (short)0, sta, false, false);
    stb = __builtin_amdgcn_wmma_f32_16x16x32_bf16(false, kb2.v, false, bq_[j].v,
                                                  (short)0, stb, false, false);
  }

  // ---- (mask +) per-lane online softmax, log2 domain
  float mx = -__builtin_inff();
#pragma unroll
  for (int r = 0; r < 8; ++r) {
    if (MASKED) {
      const int keya = kk + r + half16 * 8;
      sta[r] = (keya <= query)      ? sta[r] : -__builtin_inff();
      stb[r] = (keya + 16 <= query) ? stb[r] : -__builtin_inff();
    }
    mx = fmaxf(mx, fmaxf(sta[r], stb[r]));
  }
  mx = fmaxf(mx, __shfl_xor(mx, 16, 32));          // join the two key-halves
  const float mnew = fmaxf(m, mx);
  const float corr = __builtin_amdgcn_exp2f(m - mnew);
  m = mnew;

  float pa[8], pb[8];
  float rs = 0.0f;
#pragma unroll
  for (int r = 0; r < 8; ++r) {                    // raw v_exp_f32, mask -> 0
    pa[r] = __builtin_amdgcn_exp2f(sta[r] - mnew);
    pb[r] = __builtin_amdgcn_exp2f(stb[r] - mnew);
    rs += pa[r] + pb[r];
  }
  rs += __shfl_xor(rs, 16, 32);
  l = l * corr + rs;
#pragma unroll
  for (int c = 0; c < 4; ++c)
#pragma unroll
    for (int r = 0; r < 8; ++r) ot[c][r] *= corr;

  // ---- build P^T B-fragment (32 keys x 16 q) via xor-16 exchanges
  Frag16 bp;
#pragma unroll
  for (int r = 0; r < 8; ++r) {
    const float pax = __shfl_xor(pa[r], 16, 32);
    const float pbx = __shfl_xor(pb[r], 16, 32);
    const float lo = half16 ? pbx   : pa[r];       // elem r   : key 16*half+r
    const float hi = half16 ? pb[r] : pax;         // elem 8+r : key 16*half+8+r
    bp.h[0][r] = f2bf(lo);
    bp.h[1][r] = f2bf(hi);
  }

  // ---- O^T += V^T @ P^T
#pragma unroll
  for (int c = 0; c < 4; ++c) {
    Frag16 av;
    const unsigned short* vp =
        Vh + (size_t)(c * 16 + lr) * SEQ + kk + (half16 ? 16 : 0);
    av.h[0] = *(const ushort8*)(vp);
    av.h[1] = *(const ushort8*)(vp + 8);
    ot[c] = __builtin_amdgcn_wmma_f32_16x16x32_bf16(false, av.v, false, bp.v,
                                                    (short)0, ot[c], false, false);
  }
}

// =====================================================================
// Causal flash attention, transposed form: one wave per 16-query tile.
// Interior key blocks take the unmasked fast path; only the 1-2 blocks
// straddling the diagonal pay the compare/select cost.
// =====================================================================
__global__ void attn_kernel(const unsigned short* __restrict__ Qb,
                            const unsigned short* __restrict__ Kb,
                            const unsigned short* __restrict__ Vt,
                            float* __restrict__ outF) {
  const int lane   = threadIdx.x & 31;
  const int wave   = threadIdx.x >> 5;
  const int half16 = (lane >= 16) ? 1 : 0;
  const int lr     = lane & 15;
  const int q0 = (blockIdx.x * 4 + wave) * 16;
  const int h  = blockIdx.y;
  const int b  = blockIdx.z;

  const unsigned short* Qh = Qb + ((size_t)(b * NH + h)) * SEQ * DKH;
  const unsigned short* Kh = Kb + ((size_t)(b * NH + h)) * SEQ * DKH;
  const unsigned short* Vh = Vt + ((size_t)(b * NH + h)) * DKH * SEQ;

  // Q^T B-fragments (32dk x 16q): lane = query, table-order dk
  Frag16 bq_[2];
  {
    const unsigned short* qrow = Qh + (size_t)(q0 + lr) * DKH;
#pragma unroll
    for (int j = 0; j < 2; ++j) {
      const int kb = 32 * j + (half16 ? 8 : 0);
      bq_[j].h[0] = *(const ushort8*)(qrow + kb);
      bq_[j].h[1] = *(const ushort8*)(qrow + kb + 16);
    }
  }

  v8f ot[4];
#pragma unroll
  for (int c = 0; c < 4; ++c) ot[c] = v8f_zero();
  float m = -__builtin_inff();
  float l = 0.0f;
  const int query = q0 + lr;

  const int kend  = q0 + 16;
  // last 32-key block fully below diagonal: kk+31 <= q0 (q0 = min query)
  const int kfull = (q0 >= 31) ? (((q0 - 31) >> 5) << 5) + 32 : 0;

  int kk = 0;
  for (; kk < kfull; kk += 32)
    attn_block<false>(Kh, Vh, bq_, ot, m, l, kk, query, half16, lr);
  for (; kk < kend; kk += 32)
    attn_block<true>(Kh, Vh, bq_, ot, m, l, kk, query, half16, lr);

  // ---- normalize and emit fp32 [b][s][h*64+d], 2x b128 per dk tile
  const float inv = 1.0f / l;
  const int s = query;
  float* dbase = outF + ((size_t)(b * SEQ + s)) * D_MODEL + h * DKH + half16 * 8;
#pragma unroll
  for (int c = 0; c < 4; ++c) {
    float4 lo = { ot[c][0] * inv, ot[c][1] * inv, ot[c][2] * inv, ot[c][3] * inv };
    float4 hi = { ot[c][4] * inv, ot[c][5] * inv, ot[c][6] * inv, ot[c][7] * inv };
    float* dst = dbase + c * 16;
    *(float4*)(dst)     = lo;
    *(float4*)(dst + 4) = hi;
  }
}

// =====================================================================
extern "C" void kernel_launch(void* const* d_in, const int* in_sizes, int n_in,
                              void* d_out, int out_size, void* d_ws, size_t ws_size,
                              hipStream_t stream) {
  (void)in_sizes; (void)n_in; (void)out_size; (void)ws_size;
  const float* x  = (const float*)d_in[0];
  const float* Wq = (const float*)d_in[1];
  const float* bq = (const float*)d_in[2];
  const float* Wk = (const float*)d_in[3];
  const float* bk = (const float*)d_in[4];
  const float* Wv = (const float*)d_in[5];
  const float* bv = (const float*)d_in[6];
  const float* Wo = (const float*)d_in[7];
  const float* bo = (const float*)d_in[8];

  char* ws = (char*)d_ws;
  size_t off = 0;
  auto carve = [&](size_t bytes) -> void* {
    void* p = ws + off;
    off = (off + bytes + 255) & ~(size_t)255;
    return p;
  };
  const size_t NE_X = (size_t)MTOT * D_MODEL;       // 4,194,304
  const size_t NE_W = (size_t)D_MODEL * D_MODEL;    // 1,048,576
  unsigned short* xb    = (unsigned short*)carve(NE_X * 2);
  unsigned short* WTq   = (unsigned short*)carve(NE_W * 2);
  unsigned short* WTk   = (unsigned short*)carve(NE_W * 2);
  unsigned short* WTv   = (unsigned short*)carve(NE_W * 2);
  unsigned short* WTo   = (unsigned short*)carve(NE_W * 2);
  unsigned short* Qb    = (unsigned short*)carve(NE_X * 2);
  unsigned short* Kb    = (unsigned short*)carve(NE_X * 2);
  unsigned short* Vtb   = (unsigned short*)carve(NE_X * 2);
  float*          attnF = (float*)carve(NE_X * 4);
  unsigned short* attnB = (unsigned short*)carve(NE_X * 2);

  // 1) precision conversion / weight transpose
  cvt_f32_to_bf16<<<(int)(NE_X / 256), 256, 0, stream>>>(x, xb, (int)NE_X);
  transpose_cvt<<<(int)(NE_W / 256), 256, 0, stream>>>(Wq, WTq);
  transpose_cvt<<<(int)(NE_W / 256), 256, 0, stream>>>(Wk, WTk);
  transpose_cvt<<<(int)(NE_W / 256), 256, 0, stream>>>(Wv, WTv);
  transpose_cvt<<<(int)(NE_W / 256), 256, 0, stream>>>(Wo, WTo);

  // 2) QKV projections (WMMA), writing head-split / transposed layouts
  dim3 ggrid(D_MODEL / 64, MTOT / 128);
  gemm_wmma<1><<<ggrid, 256, 0, stream>>>(xb, WTq, bq, nullptr, Qb);
  gemm_wmma<2><<<ggrid, 256, 0, stream>>>(xb, WTk, bk, nullptr, Kb);
  gemm_wmma<3><<<ggrid, 256, 0, stream>>>(xb, WTv, bv, nullptr, Vtb);

  // 3) causal flash attention (WMMA, transposed form, split mask path)
  attn_kernel<<<dim3(SEQ / 16 / 4, NH, BATCH), 128, 0, stream>>>(Qb, Kb, Vtb, attnF);

  // 4) output projection (WMMA) -> fp32 d_out
  cvt_f32_to_bf16<<<(int)(NE_X / 256), 256, 0, stream>>>(attnF, attnB, (int)NE_X);
  gemm_wmma<0><<<ggrid, 256, 0, stream>>>(attnB, WTo, bo, (float*)d_out, nullptr);
}
